// MMoE_11836929868180
// MI455X (gfx1250) — matrix operand order repaired and checked by reading, not compiled, not run on hardware
//
#include <hip/hip_runtime.h>
#include <hip/hip_bf16.h>
#include <math.h>
#include <stdint.h>

typedef __attribute__((ext_vector_type(16))) _Float16 v16h;
typedef __attribute__((ext_vector_type(8)))  float    v8f;

#define B_TOT   16384
#define L_SEQ   200
#define NEG_BIG (-1e9f)

// ---------------- CDNA5 WMMA helpers (wave32, v_wmma_f32_16x16x32_f16) --------
__device__ __forceinline__ v8f wmma16(v16h a, v16h b, v8f c) {
  return __builtin_amdgcn_wmma_f32_16x16x32_f16(false, a, false, b, (short)0, c,
                                                false, false);
}
__device__ __forceinline__ v8f splat8(float x) {
  v8f c;
#pragma unroll
  for (int r = 0; r < 8; ++r) c[r] = x;
  return c;
}
// A fragment (16 rows x 32 K, f16) from row-major LDS tile [16][ldk].
// ISA layout: m = lane&15 ; kbase = (lane<16?0:8) ; VGPR j<4 -> k=kbase+p,
// VGPR j>=4 -> k=kbase+8+p  =>  k = kbase + p + (p>=8 ? 8 : 0)
__device__ __forceinline__ v16h load_a_frag(const _Float16* t, int ldk, int k0, int lane) {
  int m  = lane & 15;
  int kb = k0 + ((lane & 16) ? 8 : 0);
  v16h a;
#pragma unroll
  for (int p = 0; p < 16; ++p) {
    int k = kb + p + ((p >= 8) ? 8 : 0);
    a[p] = t[m * ldk + k];
  }
  return a;
}
// B fragment from pre-packed weights: frag = 32 lanes x 16 halves, per-lane contiguous.
__device__ __forceinline__ v16h load_bfrag(const _Float16* base, int frag, int lane) {
  return ((const v16h*)base)[frag * 32 + lane];
}

// ---------------- Kernel 0: pack weights into WMMA-B fragment order -----------
// src is (K x N) row-major f32 (optionally srcA + sign*srcB). Zero-pads K/N.
// frag linear index = (batch*nN + nt)*nK + ks ; within frag lane*16+p.
// B layout assumed: n = lane&15 ; k = ks*32 + (lane<16 ? 0 : 8) + p + (p>=8 ? 8 : 0).
__global__ void pack_b_kernel(const float* __restrict__ srcA, const float* __restrict__ srcB,
                              float signB, _Float16* __restrict__ dst,
                              int K, int N, int nK, int nN, long srcStride) {
  long tid  = (long)blockIdx.x * blockDim.x + threadIdx.x;
  int  p    = (int)(tid & 15);
  int  lane = (int)((tid >> 4) & 31);
  long frag = tid >> 9;
  int  ks   = (int)(frag % nK);
  long t2   = frag / nK;
  int  nt   = (int)(t2 % nN);
  long bi   = t2 / nN;
  int n = nt * 16 + (lane & 15);
  int k = ks * 32 + ((lane & 16) ? 8 : 0) + p + ((p >= 8) ? 8 : 0);
  float v = 0.f;
  if (k < K && n < N) {
    long off = bi * srcStride + (long)k * N + n;
    v = srcA[off];
    if (srcB) v += signB * srcB[off];
  }
  dst[frag * 512 + lane * 16 + p] = (_Float16)v;
}

// ---------------- Kernel 1: tgt = i_emb @ Wproj ; tb = tgt@(Wa+Wc)+ab1 --------
__global__ void encode_kernel(const int* __restrict__ item_id, const float* __restrict__ item_E,
                              const _Float16* __restrict__ Wproj_p, const _Float16* __restrict__ Wt_p,
                              const float* __restrict__ ab1,
                              _Float16* __restrict__ tgt_h, float* __restrict__ tb_ws) {
  extern __shared__ char smem[];
  int lane = threadIdx.x & 31, wid = threadIdx.x >> 5;
  _Float16* abuf = (_Float16*)smem + wid * (16 * 64 * 2);
  _Float16* tbuf = abuf + 16 * 64;
  int b0 = (blockIdx.x * (blockDim.x >> 5) + wid) * 16;

  { // gather 16 item embeddings (f32 -> f16 LDS); 2 lanes per row
    int r = lane >> 1, co = (lane & 1) * 32;
    const float* src = item_E + (long)item_id[b0 + r] * 64 + co;
    _Float16* d = abuf + r * 64 + co;
#pragma unroll
    for (int i = 0; i < 32; i += 4) {
      float4 v = *(const float4*)(src + i);
      d[i] = (_Float16)v.x; d[i + 1] = (_Float16)v.y;
      d[i + 2] = (_Float16)v.z; d[i + 3] = (_Float16)v.w;
    }
  }
  __syncthreads();
  int n = lane & 15, mb = (lane >> 4) * 8;
  v16h a0 = load_a_frag(abuf, 64, 0, lane);
  v16h a1 = load_a_frag(abuf, 64, 32, lane);
#pragma unroll
  for (int nt = 0; nt < 4; ++nt) {
    v8f c = splat8(0.f);
    c = wmma16(a0, load_bfrag(Wproj_p, nt * 2 + 0, lane), c);
    c = wmma16(a1, load_bfrag(Wproj_p, nt * 2 + 1, lane), c);
#pragma unroll
    for (int r = 0; r < 8; ++r) {
      _Float16 hv = (_Float16)c[r];
      tbuf[(mb + r) * 64 + nt * 16 + n] = hv;
      tgt_h[(long)(b0 + mb + r) * 64 + nt * 16 + n] = hv;
    }
  }
  __syncthreads();
  v16h t0 = load_a_frag(tbuf, 64, 0, lane);
  v16h t1 = load_a_frag(tbuf, 64, 32, lane);
#pragma unroll
  for (int nt = 0; nt < 4; ++nt) {
    v8f c = splat8(ab1[nt * 16 + n]);
    c = wmma16(t0, load_bfrag(Wt_p, nt * 2 + 0, lane), c);
    c = wmma16(t1, load_bfrag(Wt_p, nt * 2 + 1, lane), c);
#pragma unroll
    for (int r = 0; r < 8; ++r)
      tb_ws[(long)(b0 + mb + r) * 64 + nt * 16 + n] = c[r];
  }
}

// ---------------- Kernel 2: DIN attention with online softmax -----------------
// one wave == one batch row ; M-tile = 16 sequence positions
// per-wave LDS: hbuf32(4096) Hbuf(2048) tlf(256) sc(1024) sv(64) idxb(64)
#define ATT_WAVE_BYTES 7552
__global__ void attn_kernel(const int* __restrict__ hist, const float* __restrict__ hist_E,
                            const _Float16* __restrict__ tgt_h, const float* __restrict__ tb_ws,
                            const _Float16* __restrict__ W1p, const _Float16* __restrict__ W2p,
                            const _Float16* __restrict__ aW2p,
                            const float* __restrict__ ab2, const float* __restrict__ aW3,
                            const float* __restrict__ ab3,
                            float* __restrict__ pooled_ws) {
  extern __shared__ char smem[];
  int lane = threadIdx.x & 31, wid = threadIdx.x >> 5;
  char* wb = smem + wid * ATT_WAVE_BYTES;
  float*    hbuf32 = (float*)wb;              // 16x64 f32 (history chunk, async DMA dest)
  _Float16* Hbuf   = (_Float16*)(wb + 4096);  // 16x64 f16 (hidden1)
  float*    tlf    = (float*)(wb + 6144);     // 64 f32   (target vec)
  float*    sc     = (float*)(wb + 6400);     // 16x16 f32 (hidden2)
  float*    sv     = (float*)(wb + 7424);     // 16 f32   (chunk scores)
  int*      idxb   = (int*)(wb + 7488);       // 16 idx
  int b = blockIdx.x * (blockDim.x >> 5) + wid;

  v16h w1f[8], w2f[8], w3f[2];                // B fragments kept in VGPRs
#pragma unroll
  for (int i = 0; i < 8; ++i) { w1f[i] = load_bfrag(W1p, i, lane); w2f[i] = load_bfrag(W2p, i, lane); }
  w3f[0] = load_bfrag(aW2p, 0, lane); w3f[1] = load_bfrag(aW2p, 1, lane);

  tlf[lane]      = (float)tgt_h[(long)b * 64 + lane];
  tlf[lane + 32] = (float)tgt_h[(long)b * 64 + lane + 32];
  int n = lane & 15, mb = (lane >> 4) * 8;
  float tbv[4];
#pragma unroll
  for (int nt = 0; nt < 4; ++nt) tbv[nt] = tb_ws[(long)b * 64 + nt * 16 + n];
  float b2 = ab2[n], b3 = ab3[0];
  float m_run = -INFINITY, s_run = 0.f, pool0 = 0.f, pool1 = 0.f;
  __syncthreads();

  for (int ch = 0; ch < 13; ++ch) {      // 13*16 = 208 >= L_SEQ (padded)
    int l0 = ch * 16;
    if (lane < 16) {
      int l = l0 + lane;
      idxb[lane] = (l < L_SEQ) ? hist[(long)b * L_SEQ + l] : 0;
    }
    __builtin_prefetch(hist + (long)b * L_SEQ + l0 + 16, 0, 0);
    __syncthreads();
    { // async DMA gather of 16 history rows (f32) straight into LDS:
      // each lane owns half a row (128B); 8 x GLOBAL_LOAD_ASYNC_TO_LDS_B128
      int r = lane >> 1, co = lane & 1;
      unsigned long long gbase = (unsigned long long)(uintptr_t)hist_E +
                                 (unsigned long long)(unsigned)idxb[r] * 256ull +
                                 (unsigned)(co * 128);
      unsigned ldsb = (unsigned)(uintptr_t)hbuf32 + (unsigned)(r * 256 + co * 128);
#pragma unroll
      for (int i = 0; i < 8; ++i) {
        unsigned long long ga = gbase + (unsigned)(i * 16);
        unsigned la = ldsb + (unsigned)(i * 16);
        asm volatile("global_load_async_to_lds_b128 %0, %1, off"
                     :: "v"(la), "v"(ga) : "memory");
      }
      asm volatile("s_wait_asynccnt 0x0" ::: "memory");
    }
    __syncthreads();
    // layer1: relu(tb + h@(Wb-Wc) + (t*h)@Wd)
    v8f acc[4];
#pragma unroll
    for (int nt = 0; nt < 4; ++nt) acc[nt] = splat8(tbv[nt]);
#pragma unroll
    for (int ks = 0; ks < 2; ++ks) {
      int m = lane & 15;
      int kb = ks * 32 + ((lane & 16) ? 8 : 0);
      v16h a1, a2;
#pragma unroll
      for (int p = 0; p < 16; ++p) {
        int k = kb + p + ((p >= 8) ? 8 : 0);
        float hv = hbuf32[m * 64 + k];
        a1[p] = (_Float16)hv;
        a2[p] = (_Float16)(hv * tlf[k]);
      }
#pragma unroll
      for (int nt = 0; nt < 4; ++nt) {
        acc[nt] = wmma16(a1, w1f[nt * 2 + ks], acc[nt]);
        acc[nt] = wmma16(a2, w2f[nt * 2 + ks], acc[nt]);
      }
    }
#pragma unroll
    for (int nt = 0; nt < 4; ++nt)
#pragma unroll
      for (int r = 0; r < 8; ++r)
        Hbuf[(mb + r) * 64 + nt * 16 + n] = (_Float16)fmaxf(acc[nt][r], 0.f);
    __syncthreads();
    // layer2: relu(H @ aW2 + ab2)
    v8f c2 = splat8(b2);
    c2 = wmma16(load_a_frag(Hbuf, 64, 0, lane),  w3f[0], c2);
    c2 = wmma16(load_a_frag(Hbuf, 64, 32, lane), w3f[1], c2);
#pragma unroll
    for (int r = 0; r < 8; ++r) sc[(mb + r) * 16 + n] = fmaxf(c2[r], 0.f);
    __syncthreads();
    // layer3 + mask
    if (lane < 16) {
      float s = b3;
#pragma unroll
      for (int k = 0; k < 16; ++k) s += sc[lane * 16 + k] * aW3[k];
      int l = l0 + lane;
      if (l >= L_SEQ || idxb[lane] == 0) s = NEG_BIG;
      sv[lane] = s;
    }
    __syncthreads();
    // online softmax + pooled accumulation (each lane owns cols lane, lane+32)
    float cmax = -INFINITY;
#pragma unroll
    for (int m = 0; m < 16; ++m) cmax = fmaxf(cmax, sv[m]);
    float nm = fmaxf(m_run, cmax);
    float scale = __expf(m_run - nm);
    float es = 0.f, p0 = 0.f, p1 = 0.f;
#pragma unroll
    for (int m = 0; m < 16; ++m) {
      float e = __expf(sv[m] - nm);
      es += e;
      p0 += e * hbuf32[m * 64 + lane];
      p1 += e * hbuf32[m * 64 + lane + 32];
    }
    s_run = s_run * scale + es;
    pool0 = pool0 * scale + p0;
    pool1 = pool1 * scale + p1;
    m_run = nm;
    __syncthreads();
  }
  float inv = 1.f / s_run;
  pooled_ws[(long)b * 64 + lane]      = pool0 * inv;
  pooled_ws[(long)b * 64 + lane + 32] = pool1 * inv;
}

// ---------------- Kernel 3: experts + gates + towers --------------------------
#define K3_SMEM (8192 + 65536 + 32768 + 8192 + 4096 + 2048 + 4096)
__global__ void mmoe_kernel(const int* __restrict__ user_id, const int* __restrict__ item_id,
                            const int* __restrict__ cat_id, const int* __restrict__ dur_id,
                            const float* __restrict__ user_dense, const float* __restrict__ item_dense,
                            const float* __restrict__ user_E, const float* __restrict__ item_E,
                            const float* __restrict__ cat_E, const float* __restrict__ dur_E,
                            const float* __restrict__ pooled,
                            const _Float16* __restrict__ eW1p, const float* __restrict__ eb1,
                            const _Float16* __restrict__ eW2p, const float* __restrict__ eb2,
                            const _Float16* __restrict__ gWp, const float* __restrict__ gb,
                            const _Float16* __restrict__ tW1p, const float* __restrict__ tb1,
                            const _Float16* __restrict__ tW2p, const float* __restrict__ tb2,
                            const float* __restrict__ tW3, const float* __restrict__ tb3,
                            float* __restrict__ out) {
  extern __shared__ char smem[];
  _Float16* sh  = (_Float16*)smem;             // 16x256 shared features (K padded 244->256)
  _Float16* ehb = sh + 16 * 256;               // 8 x 16x256 expert hidden
  _Float16* eob = ehb + 8 * 16 * 256;          // 8 x 16x128 expert out
  _Float16* tib = eob + 8 * 16 * 128;          // 2 x 16x128 gated mix
  _Float16* x1b = tib + 2 * 16 * 128;          // 2 x 16x64 tower hidden1
  float*    gsc = (float*)(x1b + 2 * 16 * 64); // 2 x 16x16 gate scores
  float*    x2b = gsc + 2 * 16 * 16;           // 2 x 16x32 tower hidden2
  int tid = threadIdx.x, lane = tid & 31, wid = tid >> 5;
  int b0 = blockIdx.x * 16;

  // build shared feature tile (concat of all encoded features)
  for (int i = tid; i < 16 * 256; i += blockDim.x) {
    int m = i >> 8, col = i & 255, b = b0 + m;
    float v;
    if      (col < 64)  v = user_E[(long)user_id[b] * 64 + col];
    else if (col < 128) v = item_E[(long)item_id[b] * 64 + (col - 64)];
    else if (col < 144) v = cat_E[(long)cat_id[b] * 16 + (col - 128)];
    else if (col < 152) v = dur_E[(long)dur_id[b] * 8 + (col - 144)];
    else if (col < 177) v = user_dense[(long)b * 25 + (col - 152)];
    else if (col < 180) v = item_dense[(long)b * 3 + (col - 177)];
    else if (col < 244) v = pooled[(long)b * 64 + (col - 180)];
    else                v = 0.f;
    sh[i] = (_Float16)v;
  }
  __syncthreads();
  int n = lane & 15, mb = (lane >> 4) * 8;
  { // expert e = wave id (8 waves)
    int e = wid;
    v16h a[8];
#pragma unroll
    for (int ks = 0; ks < 8; ++ks) a[ks] = load_a_frag(sh, 256, ks * 32, lane);
    _Float16* eh = ehb + e * 16 * 256;
    for (int nt = 0; nt < 16; ++nt) {
      v8f c = splat8(eb1[e * 256 + nt * 16 + n]);
#pragma unroll
      for (int ks = 0; ks < 8; ++ks)
        c = wmma16(a[ks], load_bfrag(eW1p, (e * 16 + nt) * 8 + ks, lane), c);
#pragma unroll
      for (int r = 0; r < 8; ++r)
        eh[(mb + r) * 256 + nt * 16 + n] = (_Float16)fmaxf(c[r], 0.f);
    }
    __syncthreads();
    v16h ah[8];
#pragma unroll
    for (int ks = 0; ks < 8; ++ks) ah[ks] = load_a_frag(eh, 256, ks * 32, lane);
    _Float16* eo = eob + e * 16 * 128;
    for (int nt = 0; nt < 8; ++nt) {
      v8f c = splat8(eb2[e * 128 + nt * 16 + n]);
#pragma unroll
      for (int ks = 0; ks < 8; ++ks)
        c = wmma16(ah[ks], load_bfrag(eW2p, (e * 8 + nt) * 8 + ks, lane), c);
#pragma unroll
      for (int r = 0; r < 8; ++r)
        eo[(mb + r) * 128 + nt * 16 + n] = (_Float16)fmaxf(c[r], 0.f);
    }
  }
  __syncthreads();
  if (wid < 2) { // task t = wave id
    int t = wid;
    float* g = gsc + t * 256;
    { // gate logits via WMMA (N padded 8->16; cols >= 8 unused)
      v8f c = splat8((n < 8) ? gb[t * 8 + n] : 0.f);
#pragma unroll
      for (int ks = 0; ks < 8; ++ks)
        c = wmma16(load_a_frag(sh, 256, ks * 32, lane), load_bfrag(gWp, t * 8 + ks, lane), c);
#pragma unroll
      for (int r = 0; r < 8; ++r) g[(mb + r) * 16 + n] = c[r];
    }
    __threadfence_block();
    if (lane < 16) { // softmax over 8 experts per row
      float mx = -INFINITY;
#pragma unroll
      for (int e = 0; e < 8; ++e) mx = fmaxf(mx, g[lane * 16 + e]);
      float s = 0.f;
#pragma unroll
      for (int e = 0; e < 8; ++e) { float ev = __expf(g[lane * 16 + e] - mx); g[lane * 16 + e] = ev; s += ev; }
      float inv = 1.f / s;
#pragma unroll
      for (int e = 0; e < 8; ++e) g[lane * 16 + e] *= inv;
    }
    __threadfence_block();
    _Float16* ti = tib + t * 16 * 128;   // gated expert mix
    for (int m = 0; m < 16; ++m) {
#pragma unroll
      for (int j = 0; j < 4; ++j) {
        int col = lane + 32 * j;
        float acc = 0.f;
#pragma unroll
        for (int e = 0; e < 8; ++e)
          acc += g[m * 16 + e] * (float)eob[e * 16 * 128 + m * 128 + col];
        ti[m * 128 + col] = (_Float16)acc;
      }
    }
    __threadfence_block();
    v16h at[4];
#pragma unroll
    for (int ks = 0; ks < 4; ++ks) at[ks] = load_a_frag(ti, 128, ks * 32, lane);
    _Float16* x1 = x1b + t * 16 * 64;
#pragma unroll
    for (int nt = 0; nt < 4; ++nt) {
      v8f c = splat8(tb1[t * 64 + nt * 16 + n]);
#pragma unroll
      for (int ks = 0; ks < 4; ++ks)
        c = wmma16(at[ks], load_bfrag(tW1p, (t * 4 + nt) * 4 + ks, lane), c);
#pragma unroll
      for (int r = 0; r < 8; ++r)
        x1[(mb + r) * 64 + nt * 16 + n] = (_Float16)fmaxf(c[r], 0.f);
    }
    __threadfence_block();
    v16h ax0 = load_a_frag(x1, 64, 0, lane);
    v16h ax1 = load_a_frag(x1, 64, 32, lane);
    float* x2 = x2b + t * 16 * 32;
#pragma unroll
    for (int nt = 0; nt < 2; ++nt) {
      v8f c = splat8(tb2[t * 32 + nt * 16 + n]);
      c = wmma16(ax0, load_bfrag(tW2p, (t * 2 + nt) * 2 + 0, lane), c);
      c = wmma16(ax1, load_bfrag(tW2p, (t * 2 + nt) * 2 + 1, lane), c);
#pragma unroll
      for (int r = 0; r < 8; ++r)
        x2[(mb + r) * 32 + nt * 16 + n] = fmaxf(c[r], 0.f);
    }
    __threadfence_block();
    if (lane < 16) {
      float s = tb3[t];
#pragma unroll
      for (int k = 0; k < 32; ++k) s += x2[lane * 32 + k] * tW3[t * 32 + k];
      out[(long)t * B_TOT + b0 + lane] = 1.f / (1.f + __expf(-s));
    }
  }
}

// ---------------- Workspace layout (bytes) ------------------------------------
static constexpr long OFF_WPROJ = 0;
static constexpr long OFF_WT    = OFF_WPROJ + 8L * 1024;      //  8 frags
static constexpr long OFF_W1    = OFF_WT    + 8L * 1024;
static constexpr long OFF_W2    = OFF_W1    + 8L * 1024;
static constexpr long OFF_AW2   = OFF_W2    + 8L * 1024;      //  2 frags
static constexpr long OFF_GW    = OFF_AW2   + 2L * 1024;      // 16 frags
static constexpr long OFF_TW1   = OFF_GW    + 16L * 1024;     // 32 frags
static constexpr long OFF_TW2   = OFF_TW1   + 32L * 1024;     //  8 frags
static constexpr long OFF_EW1   = OFF_TW2   + 8L * 1024;      // 1024 frags
static constexpr long OFF_EW2   = OFF_EW1   + 1024L * 1024;   // 512 frags
static constexpr long OFF_TGT   = OFF_EW2   + 512L * 1024;    // B*64 f16
static constexpr long OFF_TB    = OFF_TGT   + (long)B_TOT * 64 * 2;
static constexpr long OFF_POOL  = OFF_TB    + (long)B_TOT * 64 * 4;

extern "C" void kernel_launch(void* const* d_in, const int* in_sizes, int n_in,
                              void* d_out, int out_size, void* d_ws, size_t ws_size,
                              hipStream_t stream) {
  (void)in_sizes; (void)n_in; (void)out_size; (void)ws_size;
  const int*   user_id    = (const int*)d_in[0];
  const int*   item_id    = (const int*)d_in[1];
  const int*   cat_id     = (const int*)d_in[2];
  const int*   dur_id     = (const int*)d_in[3];
  const float* user_dense = (const float*)d_in[4];
  const float* item_dense = (const float*)d_in[5];
  const int*   hist       = (const int*)d_in[6];
  const float* user_E     = (const float*)d_in[7];
  const float* item_E     = (const float*)d_in[8];
  const float* cat_E      = (const float*)d_in[9];
  const float* dur_E      = (const float*)d_in[10];
  const float* hist_E     = (const float*)d_in[11];
  const float* Wproj      = (const float*)d_in[12];
  const float* aW1        = (const float*)d_in[13];
  const float* ab1        = (const float*)d_in[14];
  const float* aW2        = (const float*)d_in[15];
  const float* ab2        = (const float*)d_in[16];
  const float* aW3        = (const float*)d_in[17];
  const float* ab3        = (const float*)d_in[18];
  const float* eW1        = (const float*)d_in[19];
  const float* eb1        = (const float*)d_in[20];
  const float* eW2        = (const float*)d_in[21];
  const float* eb2        = (const float*)d_in[22];
  const float* gW         = (const float*)d_in[23];
  const float* gb         = (const float*)d_in[24];
  const float* tW1        = (const float*)d_in[25];
  const float* tb1        = (const float*)d_in[26];
  const float* tW2        = (const float*)d_in[27];
  const float* tb2        = (const float*)d_in[28];
  const float* tW3        = (const float*)d_in[29];
  const float* tb3        = (const float*)d_in[30];
  float* out = (float*)d_out;
  char*  ws  = (char*)d_ws;

  _Float16* Wproj_p = (_Float16*)(ws + OFF_WPROJ);
  _Float16* Wt_p    = (_Float16*)(ws + OFF_WT);
  _Float16* W1_p    = (_Float16*)(ws + OFF_W1);
  _Float16* W2_p    = (_Float16*)(ws + OFF_W2);
  _Float16* aW2_p   = (_Float16*)(ws + OFF_AW2);
  _Float16* gW_p    = (_Float16*)(ws + OFF_GW);
  _Float16* tW1_p   = (_Float16*)(ws + OFF_TW1);
  _Float16* tW2_p   = (_Float16*)(ws + OFF_TW2);
  _Float16* eW1_p   = (_Float16*)(ws + OFF_EW1);
  _Float16* eW2_p   = (_Float16*)(ws + OFF_EW2);
  _Float16* tgt_h   = (_Float16*)(ws + OFF_TGT);
  float*    tb_ws   = (float*)(ws + OFF_TB);
  float*    pool_ws = (float*)(ws + OFF_POOL);

  hipFuncSetAttribute(reinterpret_cast<const void*>(&encode_kernel),
                      hipFuncAttributeMaxDynamicSharedMemorySize, 32768);
  hipFuncSetAttribute(reinterpret_cast<const void*>(&attn_kernel),
                      hipFuncAttributeMaxDynamicSharedMemorySize, 4 * ATT_WAVE_BYTES);
  hipFuncSetAttribute(reinterpret_cast<const void*>(&mmoe_kernel),
                      hipFuncAttributeMaxDynamicSharedMemorySize, K3_SMEM);

  auto packB = [&](const float* A, const float* Bm, float sgn, _Float16* dst,
                   int K, int N, int nK, int nN, int batch, long stride) {
    long total = (long)batch * nN * nK * 512;
    pack_b_kernel<<<(unsigned)(total / 256), 256, 0, stream>>>(A, Bm, sgn, dst, K, N, nK, nN, stride);
  };
  // aW1 blocks (rows of the 256x64): Wa=[0:64) Wb=[64:128) Wc=[128:192) Wd=[192:256)
  packB(Wproj,         nullptr,       0.f, Wproj_p, 64, 64, 2, 4, 1, 0);
  packB(aW1,           aW1 + 128*64, +1.f, Wt_p,    64, 64, 2, 4, 1, 0);  // Wa+Wc
  packB(aW1 + 64*64,   aW1 + 128*64, -1.f, W1_p,    64, 64, 2, 4, 1, 0);  // Wb-Wc
  packB(aW1 + 192*64,  nullptr,       0.f, W2_p,    64, 64, 2, 4, 1, 0);  // Wd
  packB(aW2,           nullptr,       0.f, aW2_p,   64, 16, 2, 1, 1, 0);
  packB(gW,            nullptr,       0.f, gW_p,   244,  8, 8, 1, 2, 244L * 8);
  packB(tW1,           nullptr,       0.f, tW1_p,  128, 64, 4, 4, 2, 128L * 64);
  packB(tW2,           nullptr,       0.f, tW2_p,   64, 32, 2, 2, 2, 64L * 32);
  packB(eW1,           nullptr,       0.f, eW1_p,  244, 256, 8, 16, 8, 244L * 256);
  packB(eW2,           nullptr,       0.f, eW2_p,  256, 128, 8, 8, 8, 256L * 128);

  encode_kernel<<<B_TOT / (16 * 8), 256, 32768, stream>>>(
      item_id, item_E, Wproj_p, Wt_p, ab1, tgt_h, tb_ws);
  attn_kernel<<<B_TOT / 4, 128, 4 * ATT_WAVE_BYTES, stream>>>(
      hist, hist_E, tgt_h, tb_ws, W1_p, W2_p, aW2_p, ab2, aW3, ab3, pool_ws);
  mmoe_kernel<<<B_TOT / 16, 256, K3_SMEM, stream>>>(
      user_id, item_id, cat_id, dur_id, user_dense, item_dense,
      user_E, item_E, cat_E, dur_E, pool_ws,
      eW1_p, eb1, eW2_p, eb2, gW_p, gb, tW1_p, tb1, tW2_p, tb2, tW3, tb3, out);
}
